// SCS_22866405884120
// MI455X (gfx1250) — matrix operand order repaired and checked by reading, not compiled, not run on hardware
//
#include <hip/hip_runtime.h>
#include <hip/hip_bf16.h>
#include <stdint.h>

// ---------------- problem constants ----------------
constexpr int CB   = 4;      // batches
constexpr int CN   = 2048;   // rows per batch
constexpr int CK   = 2000;   // input features
constexpr int CKP  = 2016;   // padded K (multiple of 32)
constexpr int CA0  = 256;    // W_cl rows
constexpr int CMID = 384;    // concat feature dim
constexpr int COUT = 128;    // output embedding dim
constexpr int CROWS = CB * CN; // 8192
static_assert(CK % 16 == 0 && CKP % 32 == 0 && CKP - CK < 32, "K tail assumptions");

// ---------------- workspace layout ----------------
__host__ __device__ constexpr size_t alignUp(size_t x) { return (x + 255) & ~size_t(255); }
constexpr size_t OFF_WB  = 0;                         constexpr size_t SZ_WB  = (size_t)CMID * CKP * 2;
constexpr size_t OFF_SB1 = alignUp(OFF_WB + SZ_WB);   constexpr size_t SZ_SB  = (size_t)CROWS * CMID * 2;
constexpr size_t OFF_SB2 = alignUp(OFF_SB1 + SZ_SB);
constexpr size_t OFF_CSQ = alignUp(OFF_SB2 + SZ_SB);  constexpr size_t SZ_CSQ = (size_t)2 * CB * CMID * 4;
constexpr size_t OFF_E1  = alignUp(OFF_CSQ + SZ_CSQ); constexpr size_t SZ_E   = (size_t)CB * COUT * CMID * 2;
constexpr size_t OFF_E2  = alignUp(OFF_E1 + SZ_E);
constexpr size_t OFF_YB1 = alignUp(OFF_E2 + SZ_E);    constexpr size_t SZ_YB  = (size_t)CROWS * COUT * 2;
constexpr size_t OFF_YB2 = alignUp(OFF_YB1 + SZ_YB);
constexpr size_t OFF_N1  = alignUp(OFF_YB2 + SZ_YB);  constexpr size_t SZ_N   = (size_t)CROWS * 4;
constexpr size_t OFF_N2  = alignUp(OFF_N1 + SZ_N);

// ---------------- WMMA types ----------------
typedef __attribute__((ext_vector_type(16))) __bf16 v16bf;
typedef __attribute__((ext_vector_type(8)))  float  v8f;

#define LDA 40   // LDS row pitch in bf16 elements (80B: 16B-aligned, bank-spread)

// ---------------- helpers ----------------
__device__ __forceinline__ unsigned short f2bf(float f) {
  __bf16 h = (__bf16)f;                        // hardware f32->bf16 (RNE)
  return __builtin_bit_cast(unsigned short, h);
}

#if __has_builtin(__builtin_amdgcn_global_load_async_to_lds_b128) && __has_builtin(__builtin_amdgcn_s_wait_asynccnt)
#define USE_ASYNC_LDS 1
#else
#define USE_ASYNC_LDS 0
#endif

// pointee type per hipcc diagnostic: int __attribute__((vector_size(16)))
typedef int vint4_t __attribute__((vector_size(4 * sizeof(int))));
typedef __attribute__((address_space(1))) vint4_t* gptr4_t;
typedef __attribute__((address_space(3))) vint4_t* lptr4_t;

__device__ __forceinline__ void cp16_g2l(const unsigned short* g, unsigned short* l) {
#if USE_ASYNC_LDS
  __builtin_amdgcn_global_load_async_to_lds_b128((gptr4_t)(void*)g, (lptr4_t)(void*)l, 0, 0);
#else
  *(uint4*)l = *(const uint4*)g;
#endif
}
__device__ __forceinline__ void lds_async_wait() {
#if USE_ASYNC_LDS
  __builtin_amdgcn_s_wait_asynccnt(0);
#endif
}

// copy one 128x32 bf16 tile, global row-major (gstride elems) -> LDS (pitch LDA)
__device__ __forceinline__ void copy_tile(unsigned short* sm, const unsigned short* g,
                                          int gstride, int kbase, int t) {
  int row = t >> 1, kh = t & 1;
  const unsigned short* src = g + (size_t)row * gstride + kbase + kh * 16;
  unsigned short* dst = sm + row * LDA + kh * 16;
  cp16_g2l(src, dst);
  cp16_g2l(src + 8, dst + 8);
}

__device__ __forceinline__ v16bf frag_pack(uint4 a, uint4 b) {
  union { uint4 q[2]; v16bf v; } u; u.q[0] = a; u.q[1] = b; return u.v;
}

// A fragment 16x32 bf16: lanes 0-15 row r hold k {0..7,16..23}; lanes 16-31 hold {8..15,24..31}
__device__ __forceinline__ v16bf load_frag_a(const unsigned short* sm, int row0, int lane) {
  int r = lane & 15, h = lane >> 4;
  const unsigned short* p = sm + (row0 + r) * LDA;
  return frag_pack(*(const uint4*)(p + h * 8), *(const uint4*)(p + 16 + h * 8));
}
// B fragment 32x16 bf16 (stored [n][k] rows): lane n=l&15; lanes<16 k {0..7,16..23}; lanes>=16 k {8..15,24..31}
__device__ __forceinline__ v16bf load_frag_b(const unsigned short* sm, int row0, int lane) {
  int n = lane & 15, h = lane >> 4;
  const unsigned short* p = sm + (row0 + n) * LDA;
  return frag_pack(*(const uint4*)(p + h * 8), *(const uint4*)(p + 16 + h * 8));
}

#define WMMA_BF16(A, Bf, C) \
  __builtin_amdgcn_wmma_f32_16x16x32_bf16(false, (A), false, (Bf), (short)0, (C), false, false)

// 8-WMMA step for the 4x2 wave grid (32x64 per wave)
__device__ __forceinline__ void mma_step(const unsigned short* Asm, const unsigned short* Bsm,
                                         v8f acc[2][4], int wm, int wn, int lane) {
  v16bf bfr[4];
#pragma unroll
  for (int j = 0; j < 4; j++) bfr[j] = load_frag_b(Bsm, wn * 64 + j * 16, lane);
#pragma unroll
  for (int i = 0; i < 2; i++) {
    v16bf a = load_frag_a(Asm, wm * 32 + i * 16, lane);
#pragma unroll
    for (int j = 0; j < 4; j++) acc[i][j] = WMMA_BF16(a, bfr[j], acc[i][j]);
  }
}

// ---------------- kernel 0: pack weights [W_cl;W1] -> bf16 [384][2016] ----------------
__global__ __launch_bounds__(256) void k_prepw(const float* __restrict__ Wcl,
                                               const float* __restrict__ W1,
                                               unsigned short* __restrict__ Wb) {
  int j = blockIdx.x;
  for (int k = threadIdx.x; k < CKP; k += blockDim.x) {
    float v = 0.f;
    if (k < CK) v = (j < CA0) ? Wcl[(size_t)j * CK + k] : W1[(size_t)(j - CA0) * CK + k];
    Wb[(size_t)j * CKP + k] = f2bf(v);
  }
}

// ---------------- kernel 1: zero column-sumsq accumulators ----------------
__global__ __launch_bounds__(256) void k_zero(float* __restrict__ p, int n) {
  int i = blockIdx.x * blockDim.x + threadIdx.x;
  if (i < n) p[i] = 0.f;
}

// ---------------- kernel 2: GEMM1  S = sigmoid(X @ Wb^T + bias), column sumsq ----------------
__global__ __launch_bounds__(256) void k_gemm1(const float* __restrict__ x,
                                               const unsigned short* __restrict__ Wb,
                                               const float* __restrict__ b1,
                                               unsigned short* __restrict__ Sb,
                                               float* __restrict__ csq) {
  __shared__ __align__(16) unsigned short Asm[128 * LDA];
  __shared__ __align__(16) unsigned short Bsm[128 * LDA];
  __shared__ float colsq[128];
  const int t = threadIdx.x, lane = t & 31, w = t >> 5;
  const int wm = w & 3, wn = w >> 2;                 // 4x2 wave grid -> 32x64 per wave
  const int mtile = blockIdx.x, ntile = blockIdx.y;
  v8f acc[2][4];
  for (int i = 0; i < 2; i++) for (int j = 0; j < 4; j++) for (int e = 0; e < 8; e++) acc[i][j][e] = 0.f;

  const float*          xg = x  + (size_t)mtile * 128 * CK;
  const unsigned short* wg = Wb + (size_t)ntile * 128 * CKP;

  const int row = t >> 1, kh = t & 1;
  const float* arow = xg + (size_t)row * CK;
  unsigned short* adst = Asm + row * LDA + kh * 16;

  // -------- main loop: 62 full k-steps, branch-free A path --------
  const int KFULL = CK / 32;                         // 62
  for (int kt = 0; kt < KFULL; ++kt) {
    __syncthreads();
    {
      int k0 = kt * 32 + kh * 16;
      const float4* s4 = (const float4*)(arow + k0);
      union { unsigned short s[16]; uint4 q[2]; } u;
#pragma unroll
      for (int i = 0; i < 4; i++) {
        float4 f = s4[i];
        u.s[4*i+0] = f2bf(f.x); u.s[4*i+1] = f2bf(f.y);
        u.s[4*i+2] = f2bf(f.z); u.s[4*i+3] = f2bf(f.w);
      }
      uint4* d = (uint4*)adst;
      d[0] = u.q[0]; d[1] = u.q[1];
    }
    copy_tile(Bsm, wg, CKP, kt * 32, t);             // async to LDS
    lds_async_wait();
    __syncthreads();
    mma_step(Asm, Bsm, acc, wm, wn, lane);
  }
  // -------- tail k-step (k = 1984..2015): chunk fully valid or fully zero --------
  {
    __syncthreads();
    int k0 = KFULL * 32 + kh * 16;
    union { unsigned short s[16]; uint4 q[2]; } u;
    if (k0 + 16 <= CK) {
      const float4* s4 = (const float4*)(arow + k0);
#pragma unroll
      for (int i = 0; i < 4; i++) {
        float4 f = s4[i];
        u.s[4*i+0] = f2bf(f.x); u.s[4*i+1] = f2bf(f.y);
        u.s[4*i+2] = f2bf(f.z); u.s[4*i+3] = f2bf(f.w);
      }
    } else {
#pragma unroll
      for (int i = 0; i < 16; i++) u.s[i] = 0;
    }
    uint4* d = (uint4*)adst;
    d[0] = u.q[0]; d[1] = u.q[1];
    copy_tile(Bsm, wg, CKP, KFULL * 32, t);          // Wb is zero-padded to CKP
    lds_async_wait();
    __syncthreads();
    mma_step(Asm, Bsm, acc, wm, wn, lane);
  }

  if (t < 128) colsq[t] = 0.f;
  __syncthreads();
  const int brow0 = mtile * 128, batch = mtile >> 4, jbase = ntile * 128;
#pragma unroll
  for (int i = 0; i < 2; i++)
#pragma unroll
    for (int j = 0; j < 4; j++)
#pragma unroll
      for (int e = 0; e < 8; e++) {
        int rl = wm * 32 + i * 16 + (lane >> 4) * 8 + e;
        int cl = wn * 64 + j * 16 + (lane & 15);
        int jg = jbase + cl;
        float v = acc[i][j][e] + ((jg < CA0) ? 0.f : b1[jg - CA0]);
        float s = 1.f / (1.f + __expf(-v));
        Sb[(size_t)(brow0 + rl) * CMID + jg] = f2bf(s);
        atomicAdd(&colsq[cl], s * s);
      }
  __syncthreads();
  if (t < 128) atomicAdd(&csq[batch * CMID + jbase + t], colsq[t]);
}

// ---------------- kernel 3: fold column norms + 0.1 scale into W2 (per batch) ----------------
__global__ __launch_bounds__(256) void k_fold(const float* __restrict__ W2,
                                              const float* __restrict__ csq,
                                              unsigned short* __restrict__ E) {
  int idx = blockIdx.x * blockDim.x + threadIdx.x;
  if (idx >= CB * COUT * CMID) return;
  int j = idx % CMID, o = (idx / CMID) % COUT, b = idx / (CMID * COUT);
  float nrm = sqrtf(csq[b * CMID + j]);
  float f = ((j < CA0) ? 1.0f : 0.1f) / fmaxf(nrm, 1e-12f);
  E[idx] = f2bf(W2[(size_t)o * CMID + j] * f);
}

// ---------------- kernel 4: GEMM2  Y = S @ E[b]^T + b2 ; row norms ----------------
__global__ __launch_bounds__(256) void k_gemm2(const unsigned short* __restrict__ Sb,
                                               const unsigned short* __restrict__ E,
                                               const float* __restrict__ b2,
                                               unsigned short* __restrict__ yb,
                                               float* __restrict__ nrm) {
  __shared__ __align__(16) unsigned short Asm[128 * LDA];
  __shared__ __align__(16) unsigned short Bsm[128 * LDA];
  __shared__ float rowsq[128];
  const int t = threadIdx.x, lane = t & 31, w = t >> 5;
  const int wm = w & 3, wn = w >> 2;
  const int mtile = blockIdx.x, batch = mtile >> 4;
  v8f acc[2][4];
  for (int i = 0; i < 2; i++) for (int j = 0; j < 4; j++) for (int e = 0; e < 8; e++) acc[i][j][e] = 0.f;

  const unsigned short* ag = Sb + (size_t)mtile * 128 * CMID;
  const unsigned short* bg = E  + (size_t)batch * COUT * CMID;

  for (int kt = 0; kt < CMID / 32; ++kt) {
    __syncthreads();
    copy_tile(Asm, ag, CMID, kt * 32, t);
    copy_tile(Bsm, bg, CMID, kt * 32, t);
    lds_async_wait();
    __syncthreads();
    mma_step(Asm, Bsm, acc, wm, wn, lane);
  }

  if (t < 128) rowsq[t] = 0.f;
  __syncthreads();
  const int brow0 = mtile * 128;
#pragma unroll
  for (int i = 0; i < 2; i++)
#pragma unroll
    for (int j = 0; j < 4; j++)
#pragma unroll
      for (int e = 0; e < 8; e++) {
        int rl = wm * 32 + i * 16 + (lane >> 4) * 8 + e;
        int cl = wn * 64 + j * 16 + (lane & 15);
        float v = acc[i][j][e] + b2[cl];
        yb[(size_t)(brow0 + rl) * COUT + cl] = f2bf(v);
        atomicAdd(&rowsq[rl], v * v);
      }
  __syncthreads();
  if (t < 128) nrm[brow0 + t] = sqrtf(rowsq[t]);
}

// ---------------- kernel 5: GEMM3  cos[b,m,n] = (y2 . y1) / max(|y2||y1|, 1e-8) ----------------
__global__ __launch_bounds__(256) void k_gemm3(const unsigned short* __restrict__ y2,
                                               const unsigned short* __restrict__ y1,
                                               const float* __restrict__ n2,
                                               const float* __restrict__ n1,
                                               float* __restrict__ out) {
  __shared__ __align__(16) unsigned short Asm[128 * LDA];
  __shared__ __align__(16) unsigned short Bsm[128 * LDA];
  __shared__ float sn2[128], sn1[128];
  const int t = threadIdx.x, lane = t & 31, w = t >> 5;
  const int wm = w & 3, wn = w >> 2;
  const int mtile = blockIdx.x, ntile = blockIdx.y, batch = blockIdx.z;
  v8f acc[2][4];
  for (int i = 0; i < 2; i++) for (int j = 0; j < 4; j++) for (int e = 0; e < 8; e++) acc[i][j][e] = 0.f;

  const unsigned short* ag = y2 + ((size_t)batch * CN + mtile * 128) * COUT;
  const unsigned short* bg = y1 + ((size_t)batch * CN + ntile * 128) * COUT;

  for (int kt = 0; kt < COUT / 32; ++kt) {
    __syncthreads();
    copy_tile(Asm, ag, COUT, kt * 32, t);
    copy_tile(Bsm, bg, COUT, kt * 32, t);
    lds_async_wait();
    __syncthreads();
    mma_step(Asm, Bsm, acc, wm, wn, lane);
  }

  if (t < 128) {
    sn2[t] = n2[batch * CN + mtile * 128 + t];
    sn1[t] = n1[batch * CN + ntile * 128 + t];
  }
  __syncthreads();
#pragma unroll
  for (int i = 0; i < 2; i++)
#pragma unroll
    for (int j = 0; j < 4; j++)
#pragma unroll
      for (int e = 0; e < 8; e++) {
        int rl = wm * 32 + i * 16 + (lane >> 4) * 8 + e;
        int cl = wn * 64 + j * 16 + (lane & 15);
        float den = fmaxf(sn2[rl] * sn1[cl], 1e-8f);
        out[((size_t)batch * CN + mtile * 128 + rl) * CN + ntile * 128 + cl] =
            acc[i][j][e] / den;
      }
}

// ---------------- host launcher ----------------
extern "C" void kernel_launch(void* const* d_in, const int* in_sizes, int n_in,
                              void* d_out, int out_size, void* d_ws, size_t ws_size,
                              hipStream_t stream) {
  (void)in_sizes; (void)n_in; (void)out_size; (void)ws_size;
  const float* x1  = (const float*)d_in[0];
  const float* x2  = (const float*)d_in[1];
  const float* Wcl = (const float*)d_in[2];
  const float* W1  = (const float*)d_in[3];
  const float* b1  = (const float*)d_in[4];
  const float* W2  = (const float*)d_in[5];
  const float* b2  = (const float*)d_in[6];

  char* ws = (char*)d_ws;
  unsigned short* Wb   = (unsigned short*)(ws + OFF_WB);
  unsigned short* Sb1  = (unsigned short*)(ws + OFF_SB1);
  unsigned short* Sb2  = (unsigned short*)(ws + OFF_SB2);
  float*          csq  = (float*)(ws + OFF_CSQ);
  float*          csq1 = csq;
  float*          csq2 = csq + CB * CMID;
  unsigned short* E1   = (unsigned short*)(ws + OFF_E1);
  unsigned short* E2   = (unsigned short*)(ws + OFF_E2);
  unsigned short* yb1  = (unsigned short*)(ws + OFF_YB1);
  unsigned short* yb2  = (unsigned short*)(ws + OFF_YB2);
  float*          nrm1 = (float*)(ws + OFF_N1);
  float*          nrm2 = (float*)(ws + OFF_N2);

  k_prepw<<<dim3(CMID), 256, 0, stream>>>(Wcl, W1, Wb);
  k_zero<<<dim3((2 * CB * CMID + 255) / 256), 256, 0, stream>>>(csq, 2 * CB * CMID);
  k_gemm1<<<dim3(CROWS / 128, CMID / 128), 256, 0, stream>>>(x1, Wb, b1, Sb1, csq1);
  k_gemm1<<<dim3(CROWS / 128, CMID / 128), 256, 0, stream>>>(x2, Wb, b1, Sb2, csq2);
  k_fold<<<dim3((CB * COUT * CMID + 255) / 256), 256, 0, stream>>>(W2, csq1, E1);
  k_fold<<<dim3((CB * COUT * CMID + 255) / 256), 256, 0, stream>>>(W2, csq2, E2);
  k_gemm2<<<dim3(CROWS / 128), 256, 0, stream>>>(Sb1, E1, b2, yb1, nrm1);
  k_gemm2<<<dim3(CROWS / 128), 256, 0, stream>>>(Sb2, E2, b2, yb2, nrm2);
  k_gemm3<<<dim3(CN / 128, CN / 128, CB), 256, 0, stream>>>(yb2, yb1, nrm2, nrm1, (float*)d_out);
}
